// CMLITargetLoss_6227702579750
// MI455X (gfx1250) — matrix operand order repaired and checked by reading, not compiled, not run on hardware
//
#include <hip/hip_runtime.h>

// ---------------------------------------------------------------------------
// CMLI target loss, B=64 T=197 D=768.
// Pass 1 (fused, streaming): bf16 cast of text/target + 1/||target_row|| +
//   image-MSE partial sums  (single read of image/text/target from HBM).
// Pass 2: per-batch 196x196 similarity on v_wmma_f32_16x16x32_bf16 with
//   post-hoc target-norm scaling (text normalization is argmax-invariant),
//   fused argmax -> token2patch indices.
// Pass 3: masked token MSE partials (gathered target rows, mostly L2 hits).
// Pass 4: deterministic single-block combine.
// ---------------------------------------------------------------------------

constexpr int B_   = 64;
constexpr int T_   = 197;
constexpr int D_   = 768;
constexpr int TM1  = 196;          // tokens with t >= 1
constexpr int ROWS = B_ * T_;      // 12608
constexpr int KT   = D_ / 32;      // 24 k-tiles of 32
constexpr int PREP_BLOCKS = ROWS / 8;       // 1576 (exact)
constexpr int TOK_BLOCKS  = (B_ * TM1) / 8; // 1568 (exact)

typedef __attribute__((ext_vector_type(16))) __bf16 v16bf;
typedef __attribute__((ext_vector_type(8)))  float  v8f;

union FragA { v16bf v; uint4 q[2]; };

static __device__ __forceinline__ float waveSum(float v) {
  #pragma unroll
  for (int off = 16; off; off >>= 1) v += __shfl_xor(v, off, 32);
  return v;
}

static __device__ __forceinline__ unsigned short f2bf_rne(float x) {
  unsigned int u = __float_as_uint(x);
  unsigned int r = u + 0x7FFFu + ((u >> 16) & 1u);
  return (unsigned short)(r >> 16);
}

// ---------------------------------------------------------------------------
// Kernel 1 (fused streaming pass): one wave per row (24 f32/lane per tensor).
//  - cast text/target rows to bf16 workspace (RNE)
//  - rnorm[row] = 1/||target_row||
//  - block partial sum of (image - target)^2  -> pimg[block]
// Grid is exact: 1576 blocks x 8 waves == 12608 rows, no stragglers.
// ---------------------------------------------------------------------------
__global__ void __launch_bounds__(256)
k_prep(const float* __restrict__ image, const float* __restrict__ text,
       const float* __restrict__ target,
       unsigned short* __restrict__ textbf, unsigned short* __restrict__ targbf,
       float* __restrict__ rnorm, float* __restrict__ pimg) {
  int wave = threadIdx.x >> 5, lane = threadIdx.x & 31;
  int row = blockIdx.x * 8 + wave;
  size_t base = (size_t)row * D_;
  float ss = 0.f;     // sum target^2
  float si = 0.f;     // sum (image-target)^2
  #pragma unroll 4
  for (int i = 0; i < D_ / 32; ++i) {
    int d = lane + 32 * i;
    float xt = text[base + d];
    float xg = target[base + d];
    float xi = image[base + d];
    textbf[base + d] = f2bf_rne(xt);
    targbf[base + d] = f2bf_rne(xg);
    ss += xg * xg;
    float df = xi - xg;
    si += df * df;
  }
  ss = waveSum(ss);
  si = waveSum(si);
  if (lane == 0) rnorm[row] = 1.0f / sqrtf(ss);
  __shared__ float ls[8];
  if (lane == 0) ls[wave] = si;
  __syncthreads();
  if (threadIdx.x == 0) {
    float S = 0.f;
    #pragma unroll
    for (int w = 0; w < 8; ++w) S += ls[w];
    pimg[blockIdx.x] = S;
  }
}

// ---------------------------------------------------------------------------
// Kernel 2: per-batch 196x196 similarity via v_wmma_f32_16x16x32_bf16 and
// scaled argmax.  grid = (13 t-tiles, 64 batches), 128 threads = 4 waves.
// Wave w owns n-tiles 4w .. 4w+3 (tiles 13..15 are padding, masked later).
// The K loop is straight-line: 4 unconditional WMMAs per iteration, EXEC
// stays all-ones, accumulators never move.
// ---------------------------------------------------------------------------
__global__ void __launch_bounds__(128)
k_sim_argmax(const unsigned short* __restrict__ textbf,
             const unsigned short* __restrict__ targbf,
             const float* __restrict__ rnorm,
             int* __restrict__ idx_out) {
  const int ttile = blockIdx.x;         // 0..12
  const int b     = blockIdx.y;         // 0..63
  const int lane  = threadIdx.x & 31;
  const int wave  = threadIdx.x >> 5;
  const int half  = lane >> 4;          // 0 | 1
  const int l15   = lane & 15;

  // A fragment (16x32 bf16): lanes 0-15 row M=l15 K{0..7,16..23};
  // lanes 16-31 same rows, K{8..15,24..31}.
  int tglob = 1 + ttile * 16 + l15;
  if (tglob > TM1) tglob = TM1;                       // clamp (results masked)
  const size_t abase = ((size_t)b * T_ + tglob) * (size_t)D_ + (size_t)(half * 8);

  // B fragment (32x16 bf16): lanes 0-15 col N=l15 K{0..7,16..23}; lanes 16-31 K{8..15,24..31}.
  size_t bbase[4];
  #pragma unroll
  for (int s = 0; s < 4; ++s) {
    int ntile = wave * 4 + s;                         // 0..15, uniform per wave
    int nglob = 1 + ntile * 16 + l15;
    if (nglob > TM1) nglob = TM1;                     // clamp (masked via -inf)
    bbase[s] = ((size_t)b * T_ + nglob) * (size_t)D_ + (size_t)(half * 8);
  }

  v8f acc[4];
  #pragma unroll
  for (int s = 0; s < 4; ++s)
    #pragma unroll
    for (int j = 0; j < 8; ++j) acc[s][j] = 0.0f;

  for (int kt = 0; kt < KT; ++kt) {
    const int k = kt * 32;
    FragA a;
    a.q[0] = *(const uint4*)(textbf + abase + k);
    a.q[1] = *(const uint4*)(textbf + abase + k + 16);
    #pragma unroll
    for (int s = 0; s < 4; ++s) {
      FragA bb;
      bb.q[0] = *(const uint4*)(targbf + bbase[s] + k);
      bb.q[1] = *(const uint4*)(targbf + bbase[s] + k + 16);
      acc[s] = __builtin_amdgcn_wmma_f32_16x16x32_bf16(
          false, a.v, false, bb.v, (short)0, acc[s], false, false);
    }
  }

  // Scale by 1/||target_n|| and argmax over n.
  // C layout: VGPR j, lanes 0-15 -> (M=j, N=lane); lanes 16-31 -> (M=j+8).
  const float NEGINF = -3.402823466e38f;
  float bestv[8]; int besti[8];
  #pragma unroll
  for (int j = 0; j < 8; ++j) { bestv[j] = NEGINF; besti[j] = 0x7fffffff; }

  #pragma unroll
  for (int s = 0; s < 4; ++s) {
    int ntile = wave * 4 + s;
    int nglob = 1 + ntile * 16 + l15;
    bool nvalid = (nglob <= TM1);
    float rn = rnorm[b * T_ + (nvalid ? nglob : TM1)];
    #pragma unroll
    for (int j = 0; j < 8; ++j) {
      float v  = nvalid ? acc[s][j] * rn : NEGINF;
      int  idx = nglob;
      #pragma unroll
      for (int off = 8; off >= 1; off >>= 1) {       // reduce within 16-lane half
        float ov = __shfl_xor(v, off, 16);
        int   oi = __shfl_xor(idx, off, 16);
        if (ov > v || (ov == v && oi < idx)) { v = ov; idx = oi; }
      }
      if (v > bestv[j] || (v == bestv[j] && idx < besti[j])) {
        bestv[j] = v; besti[j] = idx;
      }
    }
  }

  __shared__ float lv[4][16];
  __shared__ int   li[4][16];
  if (l15 == 0) {                                    // lanes 0 and 16
    #pragma unroll
    for (int j = 0; j < 8; ++j) {
      lv[wave][half * 8 + j] = bestv[j];
      li[wave][half * 8 + j] = besti[j];
    }
  }
  __syncthreads();
  if (threadIdx.x < 16) {
    int row = threadIdx.x;
    float v = lv[0][row]; int idx = li[0][row];
    #pragma unroll
    for (int w = 1; w < 4; ++w) {
      float ov = lv[w][row]; int oi = li[w][row];
      if (ov > v || (ov == v && oi < idx)) { v = ov; idx = oi; }
    }
    int tloc = ttile * 16 + row;
    if (tloc < TM1) idx_out[b * TM1 + tloc] = idx;   // token2patch index (full coords)
  }
}

// ---------------------------------------------------------------------------
// Kernel 3: masked token loss partials. One wave per (b, t>=1) row.
// Grid exact: 1568 blocks x 8 waves == 12544 rows.
// ---------------------------------------------------------------------------
__global__ void __launch_bounds__(256)
k_token_loss(const float* __restrict__ text, const float* __restrict__ target,
             const int* __restrict__ mask, const int* __restrict__ idx_out,
             float* __restrict__ psq, float* __restrict__ pcnt) {
  int wave = threadIdx.x >> 5, lane = threadIdx.x & 31;
  int r = blockIdx.x * 8 + wave;                     // 0 .. 12543
  int b = r / TM1, t = (r % TM1) + 1;
  int keep = (mask[b * T_ + t] == 0) ? 1 : 0;
  int idx = idx_out[r];
  size_t tb = ((size_t)b * T_ + t)   * (size_t)D_;
  size_t gb = ((size_t)b * T_ + idx) * (size_t)D_;
  float acc = 0.f;
  #pragma unroll 4
  for (int i = 0; i < D_ / 32; ++i) {
    int d = lane + 32 * i;
    float df = text[tb + d] - target[gb + d];
    acc += df * df;
  }
  acc = waveSum(acc);
  float s = keep ? acc : 0.f;
  float c = (float)keep;
  __shared__ float ls[8], lc[8];
  if (lane == 0) { ls[wave] = s; lc[wave] = c; }
  __syncthreads();
  if (threadIdx.x == 0) {
    float S = 0.f, C = 0.f;
    #pragma unroll
    for (int w = 0; w < 8; ++w) { S += ls[w]; C += lc[w]; }
    psq[blockIdx.x] = S; pcnt[blockIdx.x] = C;
  }
}

// ---------------------------------------------------------------------------
// Kernel 4: final combine (single block, deterministic).
// ---------------------------------------------------------------------------
__global__ void __launch_bounds__(256)
k_final(const float* __restrict__ text, const float* __restrict__ target,
        const float* __restrict__ psq, const float* __restrict__ pcnt, int np1,
        const float* __restrict__ pimg, int np2, float* __restrict__ out) {
  float s1 = 0.f, cnt = 0.f, simg = 0.f, s2 = 0.f;
  for (int i = threadIdx.x; i < np1; i += blockDim.x) { s1 += psq[i]; cnt += pcnt[i]; }
  for (int i = threadIdx.x; i < np2; i += blockDim.x) simg += pimg[i];
  for (int i = threadIdx.x; i < B_ * D_; i += blockDim.x) {     // CLS tokens
    int b = i / D_, d = i % D_;
    size_t e = (size_t)b * T_ * (size_t)D_ + d;
    float df = text[e] - target[e];
    s2 += df * df;
  }
  s1 = waveSum(s1); cnt = waveSum(cnt); simg = waveSum(simg); s2 = waveSum(s2);
  __shared__ float l1[8], l2[8], l3[8], l4[8];
  int wave = threadIdx.x >> 5, lane = threadIdx.x & 31;
  if (lane == 0) { l1[wave] = s1; l2[wave] = cnt; l3[wave] = simg; l4[wave] = s2; }
  __syncthreads();
  if (threadIdx.x == 0) {
    float S1 = 0.f, C = 0.f, SI = 0.f, S2 = 0.f;
    #pragma unroll
    for (int w = 0; w < 8; ++w) { S1 += l1[w]; C += l2[w]; SI += l3[w]; S2 += l4[w]; }
    float cls_loss  = S2 / (float)(B_ * D_);
    float text_loss = (S1 / (float)D_ + cls_loss) / (C + 1.0f);
    float img_loss  = SI / (float)((size_t)B_ * T_ * D_);
    out[0] = 0.5f * (text_loss + img_loss);
  }
}

// ---------------------------------------------------------------------------
extern "C" void kernel_launch(void* const* d_in, const int* in_sizes, int n_in,
                              void* d_out, int out_size, void* d_ws, size_t ws_size,
                              hipStream_t stream) {
  const float* image  = (const float*)d_in[0];
  const float* text   = (const float*)d_in[1];
  const float* target = (const float*)d_in[2];
  const int*   mask   = (const int*)d_in[3];

  char* ws = (char*)d_ws;
  const size_t SBF = (size_t)B_ * T_ * D_ * sizeof(unsigned short);  // 19.36 MB
  unsigned short* textbf = (unsigned short*)(ws);
  unsigned short* targbf = (unsigned short*)(ws + SBF);
  float* rnorm = (float*)(ws + 2 * SBF);                 // 12608 * 4 B
  int*   idx   = (int*)  (ws + 2 * SBF + 51200);         // 12544 * 4 B
  float* psq   = (float*)(ws + 2 * SBF + 102400);        // 1568
  float* pcnt  = psq + 2048;                             // 1568
  float* pimg  = pcnt + 2048;                            // 1576

  k_prep<<<PREP_BLOCKS, 256, 0, stream>>>(image, text, target,
                                          textbf, targbf, rnorm, pimg);

  dim3 g2(13, B_);
  k_sim_argmax<<<g2, 128, 0, stream>>>(textbf, targbf, rnorm, idx);

  k_token_loss<<<TOK_BLOCKS, 256, 0, stream>>>(text, target, mask, idx, psq, pcnt);
  k_final<<<1, 256, 0, stream>>>(text, target, psq, pcnt, TOK_BLOCKS,
                                 pimg, PREP_BLOCKS, (float*)d_out);
}
